// GeneratorBlock_47270410060048
// MI455X (gfx1250) — matrix operand order repaired and checked by reading, not compiled, not run on hardware
//
#include <hip/hip_runtime.h>
#include <stdint.h>

// ---------------------------------------------------------------------------
// MI455X (gfx1250, wave32) implementation of the StyleGAN2-ish GeneratorBlock.
// All channel-mixing GEMMs + 3x3 convs run on v_wmma_f32_16x16x32_f16
// (f32 accumulate, operands rounded to f16 at fragment load). The main GEMM
// stages B tiles into LDS with GLOBAL_LOAD_ASYNC_TO_LDS_B128 (ASYNCcnt),
// double buffered, when the toolchain exposes the builtin; otherwise falls
// back to vectorized float4 staging. Softmax/demod/bilinear/blur are f32 VALU.
// ---------------------------------------------------------------------------

typedef __attribute__((ext_vector_type(16))) _Float16 v16h;
typedef __attribute__((ext_vector_type(8)))  float    v8f;
typedef int v4i __attribute__((vector_size(16)));   // b128 payload type

#if defined(__has_builtin)
#if __has_builtin(__builtin_amdgcn_global_load_async_to_lds_b128)
#define HAVE_ASYNC_LDS 1
#endif
#endif
#ifndef HAVE_ASYNC_LDS
#define HAVE_ASYNC_LDS 0
#endif

__device__ __forceinline__ void wait_async0() {
#if HAVE_ASYNC_LDS
#if __has_builtin(__builtin_amdgcn_s_wait_asynccnt)
  __builtin_amdgcn_s_wait_asynccnt(0);
#else
  asm volatile("s_wait_asynccnt 0x0" ::: "memory");
#endif
#endif
}

__device__ __forceinline__ v8f wmma_f16(const v16h a, const v16h b, const v8f c) {
  // (neg_a, A, neg_b, B, c_mod, C, reuse_a, reuse_b)
  return __builtin_amdgcn_wmma_f32_16x16x32_f16(false, a, false, b, (short)0, c,
                                                false, false);
}

__device__ __forceinline__ void put4(v16h& a, int base, float4 u) {
  a[base + 0] = (_Float16)u.x;
  a[base + 1] = (_Float16)u.y;
  a[base + 2] = (_Float16)u.z;
  a[base + 3] = (_Float16)u.w;
}

// A fragment (16x32) from row-major W[M,K]: per lane two contiguous 8-float
// runs -> two pairs of b128 loads.
// Lane l (<16): row M = m0+l, elems 0..7 = K 0..7, 8..15 = K 16..23.
// Lane l+16:    row M = m0+l, elems 0..7 = K 8..15, 8..15 = K 24..31.
__device__ __forceinline__ v16h load_afrag_vec(const float* __restrict__ A,
                                               int lda, int m0, int k0, int lane) {
  const int hf = lane >> 4, r = lane & 15;
  const float* p = A + (size_t)(m0 + r) * lda + k0 + hf * 8;
  v16h a;
  put4(a, 0, *(const float4*)(p));
  put4(a, 4, *(const float4*)(p + 4));
  put4(a, 8, *(const float4*)(p + 16));
  put4(a, 12, *(const float4*)(p + 20));
  return a;
}

// Scalar A fragment (supports transA: A[k*lda+m]).
__device__ __forceinline__ v16h load_afrag_gen(const float* __restrict__ A,
                                               int lda, int m0, int k0, int lane,
                                               int transA) {
  const int hf = lane >> 4, r = lane & 15;
  const int row = m0 + r;
  v16h a;
#pragma unroll
  for (int i = 0; i < 16; ++i) {
    const int k = k0 + ((i >> 3) << 4) + (hf << 3) + (i & 7);
    const float v = transA ? A[(size_t)k * lda + row] : A[(size_t)row * lda + k];
    a[i] = (_Float16)v;
  }
  return a;
}

// B fragment (32x16) from an f32 LDS tile s[32][pitch] (K-major), cvt to f16.
// Lane l (<16): col = col0+l, K = 0..7,16..23; lane l+16: col = col0+l, K = 8..15,24..31.
__device__ __forceinline__ v16h load_bfrag_lds_f32(const float* __restrict__ s,
                                                   int pitch, int col0, int lane) {
  const int kbase = (lane >> 4) << 3;
  const int c = col0 + (lane & 15);
  v16h b;
#pragma unroll
  for (int i = 0; i < 16; ++i) b[i] = (_Float16)s[(kbase + (i & 7) + ((i >> 3) << 4)) * pitch + c];
  return b;
}

// B fragment (32x16) from an f16 LDS tile (used by the conv kernel).
__device__ __forceinline__ v16h load_bfrag_lds_f16(const _Float16* __restrict__ s,
                                                   int pitch, int col0, int lane) {
  const int kbase = (lane >> 4) << 3;
  const int c = col0 + (lane & 15);
  v16h b;
#pragma unroll
  for (int i = 0; i < 16; ++i) b[i] = s[(kbase + (i & 7) + ((i >> 3) << 4)) * pitch + c];
  return b;
}

// ---------------------------------------------------------------------------
// Generic channel-mixing GEMM: C[z] = epi(alpha * A @ B[z]) over N pixel cols.
// Block: 256 threads = 8 waves. Tile BM=64 x BN=128; wave w: mtile=w&3 (16
// rows), nhalf=w>>2 (64 cols) -> 4 v8f accumulators. B tiles are staged into
// a double-buffered f32 LDS tile by async LDS-DMA (ASYNCcnt) when available.
// Epilogue: t = alpha*acc + bias[m]; if lrelu t=lrelu(t);
//           if Res: t = Res[m,n] + gate*t;  C = t.
// ---------------------------------------------------------------------------
#define GBM 64
#define GBN 128
#define GBK 32
#define GPITCH 132  // 128 + 4 floats pad

__global__ __launch_bounds__(256) void wmma_gemm_kernel(
    const float* __restrict__ A, int lda, int transA, int64_t strideA,
    const float* __restrict__ B, int64_t strideB,
    float* __restrict__ C, int64_t strideC,
    const float* __restrict__ Res, int64_t strideRes,
    const float* __restrict__ bias,
    const float* __restrict__ gate_ptr, int gate_idx,
    float alpha, int do_lrelu, int M, int K, int N) {
  __shared__ float sB[2][GBK][GPITCH];
  const int z = blockIdx.z;
  A += (size_t)z * strideA;
  B += (size_t)z * strideB;
  C += (size_t)z * strideC;
  if (Res) Res += (size_t)z * strideRes;

  const int n0 = blockIdx.x * GBN;
  const int m0blk = blockIdx.y * GBM;
  const int tid = threadIdx.x;
  const int wave = tid >> 5, lane = tid & 31;
  const int mt = wave & 3, nh = wave >> 2;
  const int m0 = m0blk + mt * 16;

  // Stage one 32(K) x 128(N) f32 tile into LDS buffer `buf`.
  auto stage = [&](int k0, int buf) {
#pragma unroll
    for (int e = 0; e < 4; ++e) {
      const int idx = tid + 256 * e;       // 1024 float4 slots
      const int kk = idx >> 5, j4 = idx & 31;
      const float* g = B + (size_t)(k0 + kk) * N + n0 + j4 * 4;
#if HAVE_ASYNC_LDS
      __builtin_amdgcn_global_load_async_to_lds_b128(
          (__attribute__((address_space(1))) v4i*)g,
          (__attribute__((address_space(3))) v4i*)&sB[buf][kk][j4 * 4], 0, 0);
#else
      *(float4*)&sB[buf][kk][j4 * 4] = *(const float4*)g;
#endif
    }
  };

  v8f acc[4] = {};

  stage(0, 0);
  wait_async0();
  __syncthreads();

  int buf = 0;
  for (int k0 = 0; k0 < K; k0 += GBK) {
    const bool more = (k0 + GBK) < K;
    if (more) stage(k0 + GBK, buf ^ 1);  // overlap next-tile DMA with compute

    const v16h af = transA ? load_afrag_gen(A, lda, m0, k0, lane, 1)
                           : load_afrag_vec(A, lda, m0, k0, lane);
#pragma unroll
    for (int s = 0; s < 4; ++s) {
      const v16h bf =
          load_bfrag_lds_f32(&sB[buf][0][0], GPITCH, nh * 64 + s * 16, lane);
      acc[s] = wmma_f16(af, bf, acc[s]);
    }
    if (more) {
      wait_async0();
      __syncthreads();
    }
    buf ^= 1;
  }

  const float g = gate_ptr ? gate_ptr[gate_idx] : 0.0f;
  const int hf = lane >> 4, cl = lane & 15;
#pragma unroll
  for (int s = 0; s < 4; ++s) {
#pragma unroll
    for (int r = 0; r < 8; ++r) {
      const int row = m0 + hf * 8 + r;
      const int col = n0 + nh * 64 + s * 16 + cl;
      float t = alpha * acc[s][r];
      if (bias) t += bias[row];
      if (do_lrelu) t = t > 0.0f ? t : 0.2f * t;
      if (Res) t = Res[(size_t)row * N + col] + g * t;
      C[(size_t)row * N + col] = t;
    }
  }
}

// ---------------------------------------------------------------------------
// ctx[b,h,d,e] = sum_n ksm[d,n] * v[e,n]   (M=32, N=64, K=16384) per (b,h).
// One block per (b,h); 8 waves split as (mt in {0,1}) x (4 K-chunks of 4096);
// all fragment loads are contiguous-per-lane -> b128; cross-wave reduction
// through 32 KB LDS.
// ---------------------------------------------------------------------------
__global__ __launch_bounds__(256) void ctx_wmma_kernel(
    const float* __restrict__ Ksm, const float* __restrict__ V,
    float* __restrict__ CTX) {
  __shared__ float red[8][16][64];
  const int bh = blockIdx.x;
  const float* Kp = Ksm + (size_t)bh * 32 * 16384;
  const float* Vp = V + (size_t)bh * 64 * 16384;
  const int tid = threadIdx.x, wv = tid >> 5, lane = tid & 31;
  const int hf = lane >> 4, r16 = lane & 15;
  const int mt = wv & 1, chunk = wv >> 1;

  v8f acc[4] = {};
  const int kbeg = chunk * 4096;
  const int d = mt * 16 + r16;
  for (int kk = 0; kk < 4096; kk += 32) {
    const int k0 = kbeg + kk;
    v16h af;
    {
      const float* pa = Kp + (size_t)d * 16384 + k0 + hf * 8;
      put4(af, 0, *(const float4*)(pa));
      put4(af, 4, *(const float4*)(pa + 4));
      put4(af, 8, *(const float4*)(pa + 16));
      put4(af, 12, *(const float4*)(pa + 20));
    }
#pragma unroll
    for (int et = 0; et < 4; ++et) {
      const int e = et * 16 + r16;
      const float* pb = Vp + (size_t)e * 16384 + k0 + (hf << 3);
      v16h bf;
      put4(bf, 0, *(const float4*)(pb));
      put4(bf, 4, *(const float4*)(pb + 4));
      put4(bf, 8, *(const float4*)(pb + 16));
      put4(bf, 12, *(const float4*)(pb + 20));
      acc[et] = wmma_f16(af, bf, acc[et]);
    }
  }
#pragma unroll
  for (int et = 0; et < 4; ++et)
#pragma unroll
    for (int r = 0; r < 8; ++r)
      red[wv][hf * 8 + r][et * 16 + r16] = acc[et][r];
  __syncthreads();
  for (int idx = tid; idx < 2048; idx += 256) {
    const int row = idx >> 6, col = idx & 63;
    const int m = row >> 4, r = row & 15;
    const float s = red[0 + m][r][col] + red[2 + m][r][col] +
                    red[4 + m][r][col] + red[6 + m][r][col];
    CTX[(size_t)bh * 2048 + idx] = s;
  }
}

// ---------------------------------------------------------------------------
// Implicit-GEMM demodulated 3x3 conv, fused lrelu(conv + noise_branch).
// One block per (output row y, batch); BN=128 = full row -> trivial borders.
// Accumulates 9 taps x (CI/32) K-steps into persistent WMMA accumulators.
// ---------------------------------------------------------------------------
__global__ __launch_bounds__(256) void conv3x3_wmma_kernel(
    const float* __restrict__ Xin, const float* __restrict__ Wn,
    const float* __restrict__ Nres, float* __restrict__ Out, int CI) {
  __shared__ _Float16 sB[32][136];
  const int z = blockIdx.z;
  const float* Xp = Xin + (size_t)z * CI * 16384;
  const float* Np = Nres + (size_t)z * 64 * 16384;
  float* Op = Out + (size_t)z * 64 * 16384;
  const int y = blockIdx.x;  // 0..127
  const int tid = threadIdx.x, wave = tid >> 5, lane = tid & 31;
  const int mt = wave & 3, nh = wave >> 2;
  const int m0 = mt * 16;
  const int hf = lane >> 4, r16 = lane & 15;

  v8f acc[4] = {};

  for (int dy = 0; dy < 3; ++dy) {
    const int sy = y + dy - 1;
    if (sy < 0 || sy > 127) continue;  // uniform across block
    for (int dx = 0; dx < 3; ++dx) {
      for (int c0 = 0; c0 < CI; c0 += 32) {
#pragma unroll
        for (int e = 0; e < 16; ++e) {
          const int idx = tid + 256 * e;
          const int kk = idx >> 7, j = idx & 127;
          const int sx = j + dx - 1;
          const float v = (sx >= 0 && sx < 128)
                              ? Xp[(size_t)(c0 + kk) * 16384 + sy * 128 + sx]
                              : 0.0f;
          sB[kk][j] = (_Float16)v;
        }
        __syncthreads();
        v16h af;
        const int co = m0 + r16;
#pragma unroll
        for (int i = 0; i < 16; ++i) {
          const int kloc = ((i >> 3) << 4) + (hf << 3) + (i & 7);
          af[i] = (_Float16)Wn[((size_t)co * CI + (c0 + kloc)) * 9 + dy * 3 + dx];
        }
#pragma unroll
        for (int s = 0; s < 4; ++s) {
          const v16h bf = load_bfrag_lds_f16(&sB[0][0], 136, nh * 64 + s * 16, lane);
          acc[s] = wmma_f16(af, bf, acc[s]);
        }
        __syncthreads();
      }
    }
  }
#pragma unroll
  for (int s = 0; s < 4; ++s)
#pragma unroll
    for (int r = 0; r < 8; ++r) {
      const int row = m0 + hf * 8 + r;
      const int col = nh * 64 + s * 16 + r16;
      float t = acc[s][r] + Np[(size_t)row * 16384 + y * 128 + col];
      Op[(size_t)row * 16384 + y * 128 + col] = t > 0.0f ? t : 0.2f * t;
    }
}

// --------------------------- small f32 kernels -----------------------------

// softmax along n=16384 for one row per block (k-softmax).
__global__ __launch_bounds__(256) void softmax_rows_kernel(float* __restrict__ Kb) {
  __shared__ float sred[256];
  float* row = Kb + (size_t)blockIdx.x * 16384;
  const int tid = threadIdx.x;
  float m = -3.4e38f;
  for (int i = tid; i < 16384; i += 256) m = fmaxf(m, row[i]);
  sred[tid] = m;
  __syncthreads();
  for (int s = 128; s > 0; s >>= 1) {
    if (tid < s) sred[tid] = fmaxf(sred[tid], sred[tid + s]);
    __syncthreads();
  }
  m = sred[0];
  __syncthreads();
  float sum = 0.0f;
  for (int i = tid; i < 16384; i += 256) sum += __expf(row[i] - m);
  sred[tid] = sum;
  __syncthreads();
  for (int s = 128; s > 0; s >>= 1) {
    if (tid < s) sred[tid] += sred[tid + s];
    __syncthreads();
  }
  const float inv = 1.0f / sred[0];
  for (int i = tid; i < 16384; i += 256) row[i] = __expf(row[i] - m) * inv;
}

// softmax over d=32 per (b,h,pixel) (q-softmax). Coalesced: lane = pixel.
__global__ void softmax_d_kernel(float* __restrict__ Qb) {
  const int gid = blockIdx.x * blockDim.x + threadIdx.x;
  if (gid >= 32 * 16384) return;
  const int n = gid & 16383;
  const int bh = gid >> 14;
  float* base = Qb + (size_t)bh * 32 * 16384 + n;
  float vals[32];
  float m = -3.4e38f;
#pragma unroll
  for (int d = 0; d < 32; ++d) {
    vals[d] = base[(size_t)d * 16384];
    m = fmaxf(m, vals[d]);
  }
  float s = 0.0f;
#pragma unroll
  for (int d = 0; d < 32; ++d) {
    vals[d] = __expf(vals[d] - m);
    s += vals[d];
  }
  const float inv = 1.0f / s;
#pragma unroll
  for (int d = 0; d < 32; ++d) base[(size_t)d * 16384] = vals[d] * inv;
}

// bilinear x2 (half-pixel centers, edge clamp), generic over [BC,H,W].
__global__ void up2_kernel(const float* __restrict__ in, float* __restrict__ out,
                           int BC, int H, int W) {
  const int Ho = 2 * H, Wo = 2 * W;
  const size_t total = (size_t)BC * Ho * Wo;
  const size_t gid = (size_t)blockIdx.x * blockDim.x + threadIdx.x;
  if (gid >= total) return;
  const int ox = (int)(gid % Wo);
  size_t t = gid / Wo;
  const int oy = (int)(t % Ho);
  const int bc = (int)(t / Ho);
  const float* ip = in + (size_t)bc * H * W;
  const float sy = (oy + 0.5f) * 0.5f - 0.5f;
  const float sx = (ox + 0.5f) * 0.5f - 0.5f;
  const int y0 = (int)floorf(sy), x0 = (int)floorf(sx);
  const float wy = sy - y0, wx = sx - x0;
  const int y0c = max(y0, 0), y1c = min(y0 + 1, H - 1);
  const int x0c = max(x0, 0), x1c = min(x0 + 1, W - 1);
  const float v00 = ip[y0c * W + x0c], v01 = ip[y0c * W + x1c];
  const float v10 = ip[y1c * W + x0c], v11 = ip[y1c * W + x1c];
  out[gid] = (1.0f - wy) * ((1.0f - wx) * v00 + wx * v01) +
             wy * ((1.0f - wx) * v10 + wx * v11);
}

// demodulated weights with style==1: wmod = 2*W; d = rsqrt(sum(wmod^2)+eps).
__global__ __launch_bounds__(256) void demod_kernel(const float* __restrict__ W,
                                                    float* __restrict__ Wn, int CI) {
  __shared__ float sred[256];
  const int co = blockIdx.x;
  const int elems = CI * 9;
  const float* wp = W + (size_t)co * elems;
  float s = 0.0f;
  for (int i = threadIdx.x; i < elems; i += 256) {
    const float v = 2.0f * wp[i];
    s += v * v;
  }
  sred[threadIdx.x] = s;
  __syncthreads();
  for (int k = 128; k > 0; k >>= 1) {
    if (threadIdx.x < k) sred[threadIdx.x] += sred[threadIdx.x + k];
    __syncthreads();
  }
  const float d = rsqrtf(sred[0] + 1e-8f);
  for (int i = threadIdx.x; i < elems; i += 256)
    Wn[(size_t)co * elems + i] = 2.0f * wp[i] * d;
}

// style[b,f] = istyle[b] @ style_w[f]^T + style_b[f]   (4 x 64)
__global__ void style_kernel(const float* __restrict__ istyle,
                             const float* __restrict__ sw,
                             const float* __restrict__ sb,
                             float* __restrict__ style) {
  const int t = blockIdx.x * blockDim.x + threadIdx.x;
  if (t >= 256) return;
  const int b = t >> 6, f = t & 63;
  float s = sb[f];
  for (int l = 0; l < 512; ++l) s += istyle[b * 512 + l] * sw[f * 512 + l];
  style[t] = s;
}

// rgb[b,co,n] = sum_ci x[b,ci,n]*rgbw[co,ci]*(style[b,ci]+1) + prev[b,co,n]
__global__ void rgb_kernel(const float* __restrict__ X,
                           const float* __restrict__ rgbw,
                           const float* __restrict__ style,
                           const float* __restrict__ prev,
                           float* __restrict__ RGB) {
  const size_t gid = (size_t)blockIdx.x * blockDim.x + threadIdx.x;
  if (gid >= (size_t)4 * 16384) return;
  const int n = (int)(gid & 16383);
  const int b = (int)(gid >> 14);
  const float* xp = X + (size_t)b * 64 * 16384 + n;
  float a0 = 0.0f, a1 = 0.0f, a2 = 0.0f;
  for (int ci = 0; ci < 64; ++ci) {
    const float xs = xp[(size_t)ci * 16384] * (style[b * 64 + ci] + 1.0f);
    a0 += xs * rgbw[0 * 64 + ci];
    a1 += xs * rgbw[1 * 64 + ci];
    a2 += xs * rgbw[2 * 64 + ci];
  }
  RGB[((size_t)b * 3 + 0) * 16384 + n] = a0 + prev[((size_t)b * 3 + 0) * 16384 + n];
  RGB[((size_t)b * 3 + 1) * 16384 + n] = a1 + prev[((size_t)b * 3 + 1) * 16384 + n];
  RGB[((size_t)b * 3 + 2) * 16384 + n] = a2 + prev[((size_t)b * 3 + 2) * 16384 + n];
}

// depthwise [1,2,1]x[1,2,1]/16 blur with zero padding on [12,256,256].
__global__ void blur_kernel(const float* __restrict__ U, float* __restrict__ O) {
  const size_t total = (size_t)12 * 256 * 256;
  const size_t gid = (size_t)blockIdx.x * blockDim.x + threadIdx.x;
  if (gid >= total) return;
  const int x = (int)(gid & 255);
  size_t t = gid >> 8;
  const int y = (int)(t & 255);
  const int bc = (int)(t >> 8);
  const float* up = U + (size_t)bc * 65536;
  const float w1d[3] = {0.25f, 0.5f, 0.25f};
  float acc = 0.0f;
  for (int ky = 0; ky < 3; ++ky) {
    const int sy = y + ky - 1;
    if (sy < 0 || sy > 255) continue;
    for (int kx = 0; kx < 3; ++kx) {
      const int sx = x + kx - 1;
      if (sx < 0 || sx > 255) continue;
      acc += w1d[ky] * w1d[kx] * up[sy * 256 + sx];
    }
  }
  O[gid] = acc;
}

// ---------------------------------------------------------------------------
// Orchestration
// ---------------------------------------------------------------------------
static constexpr size_t NPIX = 16384;  // 128*128
static constexpr size_t OFF_XUP = 0;                                  // 4*128*NPIX
static constexpr size_t OFF_Q   = OFF_XUP + 4ull * 128 * NPIX;        // 4*256*NPIX
static constexpr size_t OFF_K   = OFF_Q + 4ull * 256 * NPIX;          // 4*256*NPIX
static constexpr size_t OFF_V   = OFF_K + 4ull * 256 * NPIX;          // 4*512*NPIX (also out_pre)
static constexpr size_t OFF_CTX = OFF_V + 4ull * 512 * NPIX;          // 32*2048
static constexpr size_t OFF_XA  = OFF_CTX + 32ull * 2048;             // 4*64*NPIX
static constexpr size_t OFF_H1  = OFF_XA + 4ull * 64 * NPIX;          // 4*128*NPIX
static constexpr size_t OFF_N1  = OFF_H1 + 4ull * 128 * NPIX;         // 4*64*NPIX
static constexpr size_t OFF_N2  = OFF_N1 + 4ull * 64 * NPIX;          // 4*64*NPIX
static constexpr size_t OFF_WN1 = OFF_N2 + 4ull * 64 * NPIX;          // 64*128*9
static constexpr size_t OFF_WN2 = OFF_WN1 + 64ull * 128 * 9;          // 64*64*9
static constexpr size_t OFF_STY = OFF_WN2 + 64ull * 64 * 9;           // 256
static constexpr size_t OFF_RGB = OFF_STY + 256;                      // 4*3*NPIX
static constexpr size_t OFF_RUP = OFF_RGB + 4ull * 3 * NPIX;          // 4*3*65536

extern "C" void kernel_launch(void* const* d_in, const int* in_sizes, int n_in,
                              void* d_out, int out_size, void* d_ws, size_t ws_size,
                              hipStream_t stream) {
  (void)in_sizes; (void)n_in; (void)out_size; (void)ws_size;
  const float* x       = (const float*)d_in[0];
  const float* prevrgb = (const float*)d_in[1];
  const float* istyle  = (const float*)d_in[2];
  const float* noise[2] = {(const float*)d_in[3], (const float*)d_in[4]};
  const float* conv1_w = (const float*)d_in[5];
  const float* conv2_w = (const float*)d_in[6];
  const float* style_w = (const float*)d_in[7];
  const float* style_b = (const float*)d_in[8];
  const float* rgb_w   = (const float*)d_in[9];
  const float* qw  = (const float*)d_in[10];
  const float* kw  = (const float*)d_in[11];
  const float* vw  = (const float*)d_in[12];
  const float* oww = (const float*)d_in[13];
  const float* obb = (const float*)d_in[14];
  const float* ga  = (const float*)d_in[15];
  const float* f1w = (const float*)d_in[16];
  const float* f1b = (const float*)d_in[17];
  const float* f2w = (const float*)d_in[18];
  const float* f2b = (const float*)d_in[19];
  const float* gf  = (const float*)d_in[20];

  float* ws = (float*)d_ws;
  float* XUP = ws + OFF_XUP;
  float* Qb  = ws + OFF_Q;
  float* Kb  = ws + OFF_K;
  float* Vb  = ws + OFF_V;   // recycled as attention out_pre
  float* CTXb = ws + OFF_CTX;
  float* XAb = ws + OFF_XA;
  float* H1b = ws + OFF_H1;
  float* N1  = ws + OFF_N1;
  float* N2  = ws + OFF_N2;
  float* WN1 = ws + OFF_WN1;
  float* WN2 = ws + OFF_WN2;
  float* STY = ws + OFF_STY;
  float* RGBb = ws + OFF_RGB;
  float* RUPb = ws + OFF_RUP;

  float* out_x = (float*)d_out;                   // [4,64,128,128]
  float* out_rgb = out_x + 4ull * 64 * NPIX;      // [4,3,256,256]

  // 1) bilinear x2 of x, weight demod, style projection
  {
    const size_t total = 4ull * 128 * 128 * 128;
    up2_kernel<<<(unsigned)((total + 255) / 256), 256, 0, stream>>>(x, XUP, 4 * 128, 64, 64);
  }
  demod_kernel<<<64, 256, 0, stream>>>(conv1_w, WN1, 128);
  demod_kernel<<<64, 256, 0, stream>>>(conv2_w, WN2, 64);
  style_kernel<<<1, 256, 0, stream>>>(istyle, style_w, style_b, STY);

  const float qscale = 0.42044820762685725f;  // 32^-0.25

  // 2) two attention+FF branches (sequential; buffers reused)
  for (int i = 0; i < 2; ++i) {
    const float* nz = noise[i];
    // Q = scale * qw @ nz   [256 x 16384] per batch
    wmma_gemm_kernel<<<dim3(128, 4, 4), 256, 0, stream>>>(
        qw + (size_t)i * 256 * 64, 64, 0, 0, nz, 64ll * 16384, Qb, 256ll * 16384,
        nullptr, 0, nullptr, nullptr, 0, qscale, 0, 256, 64, 16384);
    // K = scale * kw @ nz
    wmma_gemm_kernel<<<dim3(128, 4, 4), 256, 0, stream>>>(
        kw + (size_t)i * 256 * 64, 64, 0, 0, nz, 64ll * 16384, Kb, 256ll * 16384,
        nullptr, 0, nullptr, nullptr, 0, qscale, 0, 256, 64, 16384);
    // V = vw @ nz   [512 x 16384]
    wmma_gemm_kernel<<<dim3(128, 8, 4), 256, 0, stream>>>(
        vw + (size_t)i * 512 * 64, 64, 0, 0, nz, 64ll * 16384, Vb, 512ll * 16384,
        nullptr, 0, nullptr, nullptr, 0, 1.0f, 0, 512, 64, 16384);
    // softmax k over pixels; ctx = k @ v^T; softmax q over d
    softmax_rows_kernel<<<1024, 256, 0, stream>>>(Kb);
    ctx_wmma_kernel<<<32, 256, 0, stream>>>(Kb, Vb, CTXb);
    softmax_d_kernel<<<2048, 256, 0, stream>>>(Qb);
    // out_pre[(h,e),n] = ctx[h]^T @ qs[h]  -> overwrite Vb (batch = b*8+h)
    wmma_gemm_kernel<<<dim3(128, 1, 32), 256, 0, stream>>>(
        CTXb, 64, 1, 2048, Qb, 32ll * 16384, Vb, 64ll * 16384,
        nullptr, 0, nullptr, nullptr, 0, 1.0f, 0, 64, 32, 16384);
    // xa = nz + ga * (ow @ out_pre + ob)
    wmma_gemm_kernel<<<dim3(128, 1, 4), 256, 0, stream>>>(
        oww + (size_t)i * 64 * 512, 512, 0, 0, Vb, 512ll * 16384, XAb, 64ll * 16384,
        nz, 64ll * 16384, obb + i * 64, ga, i, 1.0f, 0, 64, 512, 16384);
    // h1 = lrelu(f1w @ xa + f1b)
    wmma_gemm_kernel<<<dim3(128, 2, 4), 256, 0, stream>>>(
        f1w + (size_t)i * 128 * 64, 64, 0, 0, XAb, 64ll * 16384, H1b, 128ll * 16384,
        nullptr, 0, f1b + i * 128, nullptr, 0, 1.0f, 1, 128, 64, 16384);
    // nout = xa + gf * (f2w @ h1 + f2b)
    wmma_gemm_kernel<<<dim3(128, 1, 4), 256, 0, stream>>>(
        f2w + (size_t)i * 64 * 128, 128, 0, 0, H1b, 128ll * 16384,
        (i == 0 ? N1 : N2), 64ll * 16384, XAb, 64ll * 16384, f2b + i * 64, gf, i,
        1.0f, 0, 64, 128, 16384);
  }

  // 3) conv1: x_mid = lrelu(conv3x3(XUP) + n1)   (x_mid reuses Qb)
  conv3x3_wmma_kernel<<<dim3(128, 1, 4), 256, 0, stream>>>(XUP, WN1, N1, Qb, 128);
  // 4) conv2: x_out = lrelu(conv3x3(x_mid) + n2) -> first output
  conv3x3_wmma_kernel<<<dim3(128, 1, 4), 256, 0, stream>>>(Qb, WN2, N2, out_x, 64);

  // 5) modulated 1x1 RGB + prev, then x2 upsample + blur -> second output
  rgb_kernel<<<(4 * 16384 + 255) / 256, 256, 0, stream>>>(out_x, rgb_w, STY, prevrgb, RGBb);
  {
    const size_t total = 12ull * 256 * 256;
    up2_kernel<<<(unsigned)((total + 255) / 256), 256, 0, stream>>>(RGBb, RUPb, 12, 128, 128);
    blur_kernel<<<(unsigned)((total + 255) / 256), 256, 0, stream>>>(RUPb, out_rgb);
  }
}
